// CUSTAttention_40381282517348
// MI455X (gfx1250) — matrix operand order repaired and checked by reading, not compiled, not run on hardware
//
#include <hip/hip_runtime.h>

typedef unsigned int   u32;
typedef unsigned short u16;
typedef __attribute__((ext_vector_type(16))) __bf16 v16bf;
typedef __attribute__((ext_vector_type(8)))  float  v8f;

#define B_N   16
#define C_N   256
#define H_N   144
#define BG_N  64          // B * ng
#define NTOK  5184        // tokens per group
#define NWIN  81
#define NCHK  81
#define MTOK  (BG_N*NTOK) // 331776

union FragB { v16bf v; u32 u[8]; };

__device__ __forceinline__ u16 f2bf(float f){
  u32 u = __builtin_bit_cast(u32, f);
  u32 r = u + 0x7FFFu + ((u >> 16) & 1u);
  return (u16)(r >> 16);
}
__device__ __forceinline__ float bf2f(u16 s){
  return __builtin_bit_cast(float, ((u32)s) << 16);
}
// A-fragment K-pair offset: VGPR j, lane-half h -> K base (ISA 7.12.2, 16-bit A 16x32)
__device__ __forceinline__ int kpairOff(int j, int half){
  return ((j & 4) << 2) + half * 8 + ((j & 3) << 1);
}
__device__ __forceinline__ int iclamp(int v, int lo, int hi){
  return v < lo ? lo : (v > hi ? hi : v);
}
__device__ __forceinline__ v8f wmma_bf16(v16bf a, v16bf b, v8f c){
  return __builtin_amdgcn_wmma_f32_16x16x32_bf16(false, a, false, b, (short)0, c, false, false);
}

// ---------------- weight f32 -> bf16 ----------------
__global__ void k_cvt_bf16(const float* __restrict__ src, u16* __restrict__ dst, int n){
  int i = blockIdx.x * blockDim.x + threadIdx.x;
  if (i < n) dst[i] = f2bf(src[i]);
}

// ---------------- x (B,C,H,W) -> xbf (Bg,N,C) bf16 ----------------
__global__ void __launch_bounds__(256) k_window_pack(const float* __restrict__ x, u16* __restrict__ xbf){
  int blk = blockIdx.x;            // bg*NTOK + n
  int c   = threadIdx.x;
  int bg  = blk / NTOK;
  int n   = blk - bg * NTOK;
  int win = n >> 6, p = n & 63;
  int gs1 = win / 9, gs2 = win - gs1 * 9;
  int g = bg & 3, b = bg >> 2;
  int h = (g >> 1) * 72 + gs1 * 8 + (p >> 3);
  int w = (g &  1) * 72 + gs2 * 8 + (p & 7);
  float v = x[(((size_t)b * C_N + c) * H_N + h) * H_N + w];
  xbf[(size_t)blk * C_N + c] = f2bf(v);
}

// ---------------- window means ----------------
__global__ void __launch_bounds__(256) k_win_mean(const u16* __restrict__ xbf, u16* __restrict__ meanbf){
  int blk = blockIdx.x;            // bg*NWIN + win
  int c = threadIdx.x;
  int bg = blk / NWIN;
  int win = blk - bg * NWIN;
  const u16* base = xbf + ((size_t)bg * NTOK + win * 64) * C_N + c;
  float s = 0.f;
  #pragma unroll 8
  for (int p = 0; p < 64; ++p) s += bf2f(base[(size_t)p * C_N]);
  meanbf[(size_t)blk * C_N + c] = f2bf(s * (1.0f / 64.0f));
}

// ---------------- sim = xg . mean^T, argmax -> assign  (WMMA 64x96x256) ----------------
__global__ void __launch_bounds__(256) k_assign(const u16* __restrict__ xbf, const u16* __restrict__ meanbf,
                                                int* __restrict__ assign){
  __shared__ u16   sMean[96 * 64];   // 96 windows x 64 channels (K-block)
  __shared__ float sSim[64][96];
  int blk = blockIdx.x;
  int bg = blk / NWIN;
  int win = blk - bg * NWIN;
  int tid = threadIdx.x, lane = tid & 31, wv = tid >> 5;
  int half = lane >> 4, l15 = lane & 15;
  int ms = (wv >> 1) * 16, nh = (wv & 1) * 48;
  const size_t arow = (size_t)bg * NTOK + win * 64;
  v8f acc[3] = {};
  for (int kb = 0; kb < 4; ++kb) {
    if (kb) __syncthreads();
    #pragma unroll
    for (int i = 0; i < 12; ++i) {             // 3072 dwords
      int lin = tid + i * 256;
      int wI = lin >> 5;
      int c2 = (lin & 31) << 1;
      u32 val = 0u;
      if (wI < NWIN)
        val = *(const u32*)&meanbf[((size_t)(bg * NWIN + wI)) * C_N + kb * 64 + c2];
      ((u32*)sMean)[lin] = val;
    }
    __syncthreads();
    #pragma unroll
    for (int ks = 0; ks < 2; ++ks) {
      int cb = kb * 64 + ks * 32;
      FragB fa;
      #pragma unroll
      for (int j = 0; j < 8; ++j)
        fa.u[j] = *(const u32*)&xbf[(arow + ms + l15) * C_N + cb + kpairOff(j, half)];
      #pragma unroll
      for (int nt = 0; nt < 3; ++nt) {
        int nw = nh + nt * 16 + l15;
        FragB fb;
        #pragma unroll
        for (int j = 0; j < 8; ++j)
          fb.u[j] = *(const u32*)&sMean[nw * 64 + ks * 32 + kpairOff(j, half)];
        acc[nt] = wmma_bf16(fa.v, fb.v, acc[nt]);
      }
    }
  }
  #pragma unroll
  for (int nt = 0; nt < 3; ++nt)
    #pragma unroll
    for (int r = 0; r < 8; ++r)
      sSim[ms + half * 8 + r][nh + nt * 16 + l15] = acc[nt][r];
  __syncthreads();
  if (tid < 64) {
    float best = -3.0e38f; int bi = 0;
    for (int k = 0; k < NWIN; ++k) {
      float v = sSim[tid][k];
      if (v > best) { best = v; bi = k; }
    }
    assign[arow + tid] = bi;
  }
}

// ---------------- stable counting sort per group (81 bins) ----------------
__global__ void __launch_bounds__(32) k_sort(const int* __restrict__ assign,
                                             int* __restrict__ sortIdx, int* __restrict__ idSorted){
  __shared__ int hist[NWIN * 32];
  int bg = blockIdx.x, t = threadIdx.x;
  const int seg = NTOK / 32;                  // 162
  const int* a = assign + (size_t)bg * NTOK;
  for (int i = t; i < NWIN * 32; i += 32) hist[i] = 0;
  __syncthreads();
  for (int i = 0; i < seg; ++i) hist[a[t * seg + i] * 32 + t]++;
  __syncthreads();
  if (t == 0) {
    int run = 0;
    for (int i = 0; i < NWIN * 32; ++i) { int tmp = hist[i]; hist[i] = run; run += tmp; }
  }
  __syncthreads();
  for (int i = 0; i < seg; ++i) {
    int orig = t * seg + i;
    int b = a[orig];
    int pos = hist[b * 32 + t]++;
    sortIdx[(size_t)bg * NTOK + pos] = orig;
    idSorted[(size_t)bg * NTOK + pos] = b;
  }
}

// ---------------- gather x_sorted ----------------
__global__ void __launch_bounds__(128) k_gather(const u16* __restrict__ xbf, const int* __restrict__ sortIdx,
                                                u16* __restrict__ xs){
  int row = blockIdx.x;
  int bg = row / NTOK;
  int orig = sortIdx[row];
  ((u32*)xs)[(size_t)row * 128 + threadIdx.x] =
      ((const u32*)xbf)[((size_t)bg * NTOK + orig) * 128 + threadIdx.x];
}

// ---------------- WMMA linear: out = A @ W^T + b ; EPI 0=bf16, 1=sigmoid->bf16, 2=scatter f32 ----------------
template<int EPI>
__global__ void __launch_bounds__(256) k_linear(const u16* __restrict__ A, const u16* __restrict__ W,
                                                const float* __restrict__ bias,
                                                u16* __restrict__ outB, float* __restrict__ outF,
                                                const int* __restrict__ sortIdx){
  __shared__ u16 sA[128 * 64];
  __shared__ u16 sW[64 * 64];
  int mtile = (blockIdx.x >> 2) * 128;
  int ntile = (blockIdx.x & 3) * 64;
  int tid = threadIdx.x, lane = tid & 31, wv = tid >> 5;
  int half = lane >> 4, l15 = lane & 15;
  int mstrip = wv * 16;
  v8f acc[4] = {};
  for (int kb = 0; kb < 4; ++kb) {
    if (kb) __syncthreads();
    #pragma unroll
    for (int i = 0; i < 16; ++i) {            // A tile: 4096 dwords
      int lin = tid + i * 256;
      int row = lin >> 5, c2 = (lin & 31) << 1;
      ((u32*)sA)[lin] = *(const u32*)&A[((size_t)(mtile + row)) * C_N + kb * 64 + c2];
    }
    #pragma unroll
    for (int i = 0; i < 8; ++i) {             // W tile: 2048 dwords
      int lin = tid + i * 256;
      int row = lin >> 5, c2 = (lin & 31) << 1;
      ((u32*)sW)[lin] = *(const u32*)&W[((size_t)(ntile + row)) * C_N + kb * 64 + c2];
    }
    __syncthreads();
    #pragma unroll
    for (int ks = 0; ks < 2; ++ks) {
      FragB fa;
      #pragma unroll
      for (int j = 0; j < 8; ++j)
        fa.u[j] = *(const u32*)&sA[(mstrip + l15) * 64 + ks * 32 + kpairOff(j, half)];
      #pragma unroll
      for (int nt = 0; nt < 4; ++nt) {
        FragB fb;
        #pragma unroll
        for (int j = 0; j < 8; ++j)
          fb.u[j] = *(const u32*)&sW[(nt * 16 + l15) * 64 + ks * 32 + kpairOff(j, half)];
        acc[nt] = wmma_bf16(fa.v, fb.v, acc[nt]);
      }
    }
  }
  #pragma unroll
  for (int nt = 0; nt < 4; ++nt) {
    int n = ntile + nt * 16 + l15;
    float bb = bias[n];
    #pragma unroll
    for (int r = 0; r < 8; ++r) {
      int m = mtile + mstrip + half * 8 + r;
      float v = acc[nt][r] + bb;
      if (EPI == 0) {
        outB[(size_t)m * C_N + n] = f2bf(v);
      } else if (EPI == 1) {
        outB[(size_t)m * C_N + n] = f2bf(1.0f / (1.0f + __expf(-v)));
      } else {
        int bg = m / NTOK;
        int orig = sortIdx[m];                 // inverse permutation by direct scatter
        int win = orig >> 6, p = orig & 63;
        int gs1 = win / 9, gs2 = win - gs1 * 9;
        int h = ((bg >> 1) & 1) * 72 + gs1 * 8 + (p >> 3);
        int w = (bg & 1) * 72 + gs2 * 8 + (p & 7);
        int b = bg >> 2;
        outF[(((size_t)b * C_N + n) * H_N + h) * H_N + w] = v;
      }
    }
  }
}

// ---------------- attention per (group, chunk): softmax(mask(QK^T/16)) V * gate ----------------
__global__ void __launch_bounds__(256) k_attention(const u16* __restrict__ Q, const u16* __restrict__ K,
                                                   const u16* __restrict__ V, const u16* __restrict__ G,
                                                   const int* __restrict__ idSorted, u16* __restrict__ ctx){
  __shared__ float sS[64][132];
  __shared__ u16   sP[64][128];
  __shared__ int   sQid[64];
  __shared__ int   sKid[128];
  __shared__ float sRed[64][4];
  int chunk = blockIdx.x;
  int bg = chunk / NCHK;
  int ci = chunk - bg * NCHK;
  int tid = threadIdx.x, lane = tid & 31, wv = tid >> 5;
  int half = lane >> 4, l15 = lane & 15;
  size_t qtok0 = (size_t)bg * NTOK + ci * 64;
  if (tid < 128) {
    int spos = ci * 64 - 32 + tid;
    sKid[tid] = (spos >= 0 && spos < NTOK) ? idSorted[(size_t)bg * NTOK + spos] : -1;
  } else if (tid < 192) {
    sQid[tid - 128] = idSorted[qtok0 + (tid - 128)];
  }
  { // phase 1: S = Q K^T * scale
    int ms = (wv >> 1) * 16, nh = (wv & 1) * 64;
    size_t rowB[4];
    #pragma unroll
    for (int nt = 0; nt < 4; ++nt) {
      int key = nh + nt * 16 + l15;
      int spos = iclamp(ci * 64 - 32 + key, 0, NTOK - 1);   // clamp; mask kills it
      rowB[nt] = ((size_t)bg * NTOK + spos) * C_N;
    }
    v8f acc[4] = {};
    #pragma unroll
    for (int ks = 0; ks < 8; ++ks) {
      int cb = ks * 32;
      FragB fa;
      #pragma unroll
      for (int j = 0; j < 8; ++j)
        fa.u[j] = *(const u32*)&Q[(qtok0 + ms + l15) * C_N + cb + kpairOff(j, half)];
      #pragma unroll
      for (int nt = 0; nt < 4; ++nt) {
        FragB fb;
        #pragma unroll
        for (int j = 0; j < 8; ++j)
          fb.u[j] = *(const u32*)&K[rowB[nt] + cb + kpairOff(j, half)];
        acc[nt] = wmma_bf16(fa.v, fb.v, acc[nt]);
      }
    }
    #pragma unroll
    for (int nt = 0; nt < 4; ++nt)
      #pragma unroll
      for (int r = 0; r < 8; ++r)
        sS[ms + half * 8 + r][nh + nt * 16 + l15] = acc[nt][r] * 0.0625f;  // 256^-0.5
  }
  __syncthreads();
  { // phase 2: mask + softmax (f32 in LDS), emit bf16 probs
    int row = tid >> 2, part = tid & 3, cb = part * 32;
    int qid = sQid[row];
    float mx = -3.0e38f;
    #pragma unroll 4
    for (int c = 0; c < 32; ++c) {
      int col = cb + c;
      float v = (qid == sKid[col]) ? sS[row][col] : -10000.0f;
      sS[row][col] = v;
      mx = fmaxf(mx, v);
    }
    sRed[row][part] = mx;
    __syncthreads();
    float rmax = fmaxf(fmaxf(sRed[row][0], sRed[row][1]), fmaxf(sRed[row][2], sRed[row][3]));
    float sum = 0.f;
    #pragma unroll 4
    for (int c = 0; c < 32; ++c) {
      float e = __expf(sS[row][cb + c] - rmax);
      sS[row][cb + c] = e;
      sum += e;
    }
    __syncthreads();
    sRed[row][part] = sum;
    __syncthreads();
    float inv = 1.0f / (sRed[row][0] + sRed[row][1] + sRed[row][2] + sRed[row][3]);
    #pragma unroll 4
    for (int c = 0; c < 32; ++c)
      sP[row][cb + c] = f2bf(sS[row][cb + c] * inv);
  }
  __syncthreads();
  { // phase 3: ctx = (P V) * gate
    int ms = (wv >> 1) * 16, nch = (wv & 1) * 128;
    v8f acc[8] = {};
    #pragma unroll
    for (int ks = 0; ks < 4; ++ks) {
      int kb0 = ks * 32;
      FragB fa;
      #pragma unroll
      for (int j = 0; j < 8; ++j)
        fa.u[j] = *(const u32*)&sP[ms + l15][kb0 + kpairOff(j, half)];
      #pragma unroll
      for (int nt = 0; nt < 8; ++nt) {
        int ch = nch + nt * 16 + l15;
        FragB fb;
        #pragma unroll
        for (int j = 0; j < 8; ++j) {
          int k0 = kb0 + kpairOff(j, half);
          int s0 = iclamp(ci * 64 - 32 + k0,     0, NTOK - 1);
          int s1 = iclamp(ci * 64 - 32 + k0 + 1, 0, NTOK - 1);
          u32 lo = V[((size_t)bg * NTOK + s0) * C_N + ch];
          u32 hi = V[((size_t)bg * NTOK + s1) * C_N + ch];
          fb.u[j] = lo | (hi << 16);
        }
        acc[nt] = wmma_bf16(fa.v, fb.v, acc[nt]);
      }
    }
    #pragma unroll
    for (int nt = 0; nt < 8; ++nt) {
      int ch = nch + nt * 16 + l15;
      #pragma unroll
      for (int r = 0; r < 8; ++r) {
        size_t tok = qtok0 + ms + half * 8 + r;
        float gv = bf2f(G[tok * C_N + ch]);
        ctx[tok * C_N + ch] = f2bf(acc[nt][r] * gv);
      }
    }
  }
}

extern "C" void kernel_launch(void* const* d_in, const int* in_sizes, int n_in,
                              void* d_out, int out_size, void* d_ws, size_t ws_size,
                              hipStream_t stream) {
  (void)in_sizes; (void)n_in; (void)out_size; (void)ws_size;
  const float* x   = (const float*)d_in[0];
  const float* Wq  = (const float*)d_in[1];
  const float* bq  = (const float*)d_in[2];
  const float* Wk  = (const float*)d_in[3];
  const float* bk  = (const float*)d_in[4];
  const float* Wv  = (const float*)d_in[5];
  const float* bv  = (const float*)d_in[6];
  const float* Wp  = (const float*)d_in[7];
  const float* bp  = (const float*)d_in[8];
  const float* Wg  = (const float*)d_in[9];
  const float* bgt = (const float*)d_in[10];
  float* out = (float*)d_out;

  char* ws = (char*)d_ws;
  const size_t S1 = (size_t)MTOK * C_N * sizeof(u16);       // 169,869,312 B
  u16* xbf = (u16*)(ws);            // reused as Q after gather
  u16* xs  = (u16*)(ws + S1);       // reused as ctx after gate GEMM
  u16* Kb  = (u16*)(ws + 2 * S1);
  u16* Vb  = (u16*)(ws + 3 * S1);
  u16* Gb  = (u16*)(ws + 4 * S1);
  char* p  = ws + 5 * S1;
  u16* meanbf  = (u16*)p; p += (size_t)BG_N * NWIN * C_N * sizeof(u16);
  int* assignB = (int*)p; p += (size_t)MTOK * sizeof(int);
  int* sortIdx = (int*)p; p += (size_t)MTOK * sizeof(int);
  int* idSorted= (int*)p; p += (size_t)MTOK * sizeof(int);
  u16* Wqb = (u16*)p; p += 65536 * sizeof(u16);
  u16* Wkb = (u16*)p; p += 65536 * sizeof(u16);
  u16* Wvb = (u16*)p; p += 65536 * sizeof(u16);
  u16* Wgb = (u16*)p; p += 65536 * sizeof(u16);
  u16* Wpb = (u16*)p; p += 65536 * sizeof(u16);
  u16* Qb  = xbf;
  u16* ctx = xs;

  k_cvt_bf16<<<256, 256, 0, stream>>>(Wq, Wqb, 65536);
  k_cvt_bf16<<<256, 256, 0, stream>>>(Wk, Wkb, 65536);
  k_cvt_bf16<<<256, 256, 0, stream>>>(Wv, Wvb, 65536);
  k_cvt_bf16<<<256, 256, 0, stream>>>(Wg, Wgb, 65536);
  k_cvt_bf16<<<256, 256, 0, stream>>>(Wp, Wpb, 65536);

  k_window_pack<<<MTOK, 256, 0, stream>>>(x, xbf);
  k_win_mean<<<BG_N * NWIN, 256, 0, stream>>>(xbf, meanbf);
  k_assign<<<BG_N * NWIN, 256, 0, stream>>>(xbf, meanbf, assignB);
  k_sort<<<BG_N, 32, 0, stream>>>(assignB, sortIdx, idSorted);
  k_gather<<<MTOK, 128, 0, stream>>>(xbf, sortIdx, xs);

  dim3 lgrid((MTOK / 128) * 4);
  k_linear<0><<<lgrid, 256, 0, stream>>>(xs, Wqb, bq,  Qb, nullptr, nullptr);
  k_linear<0><<<lgrid, 256, 0, stream>>>(xs, Wkb, bk,  Kb, nullptr, nullptr);
  k_linear<0><<<lgrid, 256, 0, stream>>>(xs, Wvb, bv,  Vb, nullptr, nullptr);
  k_linear<1><<<lgrid, 256, 0, stream>>>(xs, Wgb, bgt, Gb, nullptr, nullptr);

  k_attention<<<BG_N * NCHK, 256, 0, stream>>>(Qb, Kb, Vb, Gb, idSorted, ctx);

  k_linear<2><<<lgrid, 256, 0, stream>>>(ctx, Wpb, bp, nullptr, out, sortIdx);
}
